// BNNLinear_37434934952096
// MI455X (gfx1250) — 8-bit matrix operand order repaired and checked by reading, compiled, not run on hardware
//
#include <hip/hip_runtime.h>
#include <stdint.h>

typedef __attribute__((ext_vector_type(16))) int   v16i;
typedef __attribute__((ext_vector_type(8)))  float v8f;

#define GM     8192
#define GN     2048
#define GK     2048
#define TILE   128
#define KSTEP  128
#define PITCH  144
#define ABYTES (TILE * PITCH)
#define BUFBYTES (2 * ABYTES)
#define BN_EPS 1e-5

__device__ __forceinline__ void async_ld16(uint32_t lds_off, uint64_t gaddr) {
  asm volatile("global_load_async_to_lds_b128 %0, %1, off"
               :: "v"(lds_off), "v"(gaddr)
               : "memory");
}
__device__ __forceinline__ void wait_async_le8() {
  asm volatile("s_wait_asynccnt 0x8" ::: "memory");
}
__device__ __forceinline__ void wait_async_0() {
  asm volatile("s_wait_asynccnt 0x0" ::: "memory");
}

__global__ void k_binarize_fp8(const float* __restrict__ in,
                               uint32_t* __restrict__ out, int n4) {
  int i = blockIdx.x * blockDim.x + threadIdx.x;
  if (i >= n4) return;
  float4 v = ((const float4*)in)[i];
  uint32_t r;
  r  = (uint32_t)((v.x > 0.f) ? 0x38u : ((v.x < 0.f) ? 0xB8u : 0x00u));
  r |= (uint32_t)((v.y > 0.f) ? 0x38u : ((v.y < 0.f) ? 0xB8u : 0x00u)) << 8;
  r |= (uint32_t)((v.z > 0.f) ? 0x38u : ((v.z < 0.f) ? 0xB8u : 0x00u)) << 16;
  r |= (uint32_t)((v.w > 0.f) ? 0x38u : ((v.w < 0.f) ? 0xB8u : 0x00u)) << 24;
  out[i] = r;
}

__global__ void k_zero_stats(int* __restrict__ psum,
                             unsigned long long* __restrict__ psq, int n) {
  int i = blockIdx.x * blockDim.x + threadIdx.x;
  if (i < n) { psum[i] = 0; psq[i] = 0ull; }
}

__global__ __launch_bounds__(256) void k_gemm_fp8(
    const uint8_t* __restrict__ A8, const uint8_t* __restrict__ B8,
    float* __restrict__ C, int* __restrict__ psum,
    unsigned long long* __restrict__ psq) {
  extern __shared__ __align__(16) uint8_t lds[];

  const int tid  = threadIdx.x;
  const int lane = tid & 31;
  const int l15  = lane & 15;
  const int lh   = lane >> 4;
  const int wave = tid >> 5;
  const int wm   = (wave & 3) * 32;
  const int wn   = (wave >> 2) * 64;
  const int bm   = blockIdx.y * TILE;
  const int bn   = blockIdx.x * TILE;

  const int row0 = tid >> 3;
  const int seg  = (tid & 7) * 16;
  const uint32_t lbase = (uint32_t)(uintptr_t)lds;

  v8f acc[2][4];
#pragma unroll
  for (int mi = 0; mi < 2; ++mi)
#pragma unroll
    for (int ni = 0; ni < 4; ++ni)
      acc[mi][ni] = (v8f)(0.0f);

  auto issue = [&](int kt, int buf) {
    const uint32_t boff = lbase + (uint32_t)buf * BUFBYTES;
    const int k0 = kt * KSTEP;
#pragma unroll
    for (int c = 0; c < 4; ++c) {
      int row = row0 + c * 32;
      uint32_t l = boff + (uint32_t)(row * PITCH + seg);
      async_ld16(l,
                 (uint64_t)(A8 + (size_t)(bm + row) * GK + k0 + seg));
      async_ld16(l + ABYTES,
                 (uint64_t)(B8 + (size_t)(bn + row) * GK + k0 + seg));
    }
  };

  const int KT = GK / KSTEP;
  issue(0, 0);

  for (int kt = 0; kt < KT; ++kt) {
    const int cur = kt & 1;
    if (kt + 1 < KT) {
      issue(kt + 1, cur ^ 1);
      wait_async_le8();
    } else {
      wait_async_0();
    }
    __syncthreads();

    const uint8_t* ldsA = lds + cur * BUFBYTES;
    const uint8_t* ldsB = ldsA + ABYTES;

    v16i afrag[2];
#pragma unroll
    for (int mi = 0; mi < 2; ++mi) {
      const uint8_t* p = ldsA + (wm + mi * 16 + l15) * PITCH + lh * 8;
      int2* d = (int2*)&afrag[mi];
#pragma unroll
      for (int q = 0; q < 8; ++q) d[q] = *(const int2*)(p + q * 16);
    }
#pragma unroll
    for (int ni = 0; ni < 4; ++ni) {
      v16i bfrag;
      const uint8_t* p = ldsB + (wn + ni * 16 + l15) * PITCH + lh * 8;
      int2* d = (int2*)&bfrag;
#pragma unroll
      for (int q = 0; q < 8; ++q) d[q] = *(const int2*)(p + q * 16);
#pragma unroll
      for (int mi = 0; mi < 2; ++mi)
        acc[mi][ni] = __builtin_amdgcn_wmma_f32_16x16x128_fp8_fp8(
            afrag[mi], bfrag, (short)0, acc[mi][ni], false, false);
    }
    __syncthreads();
  }

#pragma unroll
  for (int mi = 0; mi < 2; ++mi)
#pragma unroll
    for (int ni = 0; ni < 4; ++ni) {
      int col = bn + wn + ni * 16 + l15;
#pragma unroll
      for (int v = 0; v < 8; ++v) {
        int row = bm + wm + mi * 16 + lh * 8 + v;
        C[(uint32_t)(row * GN + col)] = acc[mi][ni][v];
      }
    }

#pragma unroll
  for (int ni = 0; ni < 4; ++ni) {
    int s = 0, q = 0;
#pragma unroll
    for (int mi = 0; mi < 2; ++mi)
#pragma unroll
      for (int v = 0; v < 8; ++v) {
        int vi = (int)acc[mi][ni][v];
        s += vi;
        q += vi * vi;
      }
    s += __shfl_xor(s, 16, 32);
    q += __shfl_xor(q, 16, 32);
    if (lane < 16) {
      int col = bn + wn + ni * 16 + l15;
      atomicAdd(&psum[col], s);
      atomicAdd(&psq[col], (unsigned long long)(unsigned int)q);
    }
  }
}

__global__ void k_colstats_final(const int* __restrict__ psum,
                                 const unsigned long long* __restrict__ psq,
                                 const float* __restrict__ gamma,
                                 const float* __restrict__ beta,
                                 float* __restrict__ scale,
                                 float* __restrict__ bias,
                                 int N, int Bdim) {
  int col = blockIdx.x * blockDim.x + threadIdx.x;
  if (col >= N) return;
  double mean = (double)psum[col] / (double)Bdim;
  double var  = (double)psq[col] / (double)Bdim - mean * mean;
  float  sc   = gamma[col] * (float)rsqrt(var + (double)BN_EPS);
  scale[col]  = sc;
  bias[col]   = beta[col] - (float)mean * sc;
}

__global__ void k_bn_apply(float* __restrict__ out,
                           const float* __restrict__ scale,
                           const float* __restrict__ bias,
                           int N, long total4) {
  long i = (long)blockIdx.x * blockDim.x + threadIdx.x;
  if (i >= total4) return;
  float4 v = ((const float4*)out)[i];
  int c0 = (int)((i * 4) % (long)N);
  v.x = v.x * scale[c0 + 0] + bias[c0 + 0];
  v.y = v.y * scale[c0 + 1] + bias[c0 + 1];
  v.z = v.z * scale[c0 + 2] + bias[c0 + 2];
  v.w = v.w * scale[c0 + 3] + bias[c0 + 3];
  ((float4*)out)[i] = v;
}

extern "C" void kernel_launch(void* const* d_in, const int* in_sizes, int n_in,
                              void* d_out, int out_size, void* d_ws, size_t ws_size,
                              hipStream_t stream) {
  const float* x     = (const float*)d_in[0];
  const float* w     = (const float*)d_in[1];
  const float* gamma = (const float*)d_in[2];
  const float* beta  = (const float*)d_in[3];
  float* out = (float*)d_out;

  const int OUTN = in_sizes[2];
  const int K    = in_sizes[1] / OUTN;
  const int Bdim = in_sizes[0] / K;

  uint8_t* ws = (uint8_t*)d_ws;
  uint8_t* xb = ws;
  uint8_t* wb = xb + (size_t)Bdim * K;
  int* psum = (int*)(wb + (size_t)OUTN * K);
  unsigned long long* psq =
      (unsigned long long*)((uint8_t*)psum + (size_t)OUTN * 16);
  float* scale = (float*)((uint8_t*)psq + (size_t)OUTN * 8);
  float* bias  = scale + OUTN;

  int n4x = (Bdim * K) / 4;
  k_binarize_fp8<<<(n4x + 255) / 256, 256, 0, stream>>>(x, (uint32_t*)xb, n4x);
  int n4w = (OUTN * K) / 4;
  k_binarize_fp8<<<(n4w + 255) / 256, 256, 0, stream>>>(w, (uint32_t*)wb, n4w);

  k_zero_stats<<<(OUTN + 255) / 256, 256, 0, stream>>>(psum, psq, OUTN);

  dim3 gg(OUTN / TILE, Bdim / TILE);
  k_gemm_fp8<<<gg, 256, 2 * BUFBYTES, stream>>>(xb, wb, out, psum, psq);

  k_colstats_final<<<(OUTN + 255) / 256, 256, 0, stream>>>(
      psum, psq, gamma, beta, scale, bias, OUTN, Bdim);

  long total4 = (long)Bdim * OUTN / 4;
  k_bn_apply<<<(int)((total4 + 255) / 256), 256, 0, stream>>>(
      out, scale, bias, OUTN, total4);
}
